// MonoAPPNPModel_88304527606396
// MI455X (gfx1250) — matrix operand order repaired and checked by reading, not compiled, not run on hardware
//
#include <hip/hip_runtime.h>

// ---------------------------------------------------------------------------
// APPNP GNN on MI455X (gfx1250): bf16 WMMA MLP encoder + CSR-based atomic-free
// propagation + warp-reduced log_softmax.
// ---------------------------------------------------------------------------

typedef __attribute__((ext_vector_type(16))) __bf16 v16bf;
typedef __attribute__((ext_vector_type(8)))  __bf16 bf16x8;
typedef __attribute__((ext_vector_type(8)))  float  v8f;

#define F_IN   512
#define HID    256
#define NCLS   64
#define ALPHA  0.1f

// -------------------------------- setup kernels ----------------------------

__global__ void init_counts_kernel(int* __restrict__ counts, int n) {
    int i = blockIdx.x * blockDim.x + threadIdx.x;
    if (i < n) counts[i] = 0;
}

// Transpose + convert weights to bf16: w1t[n,k] = W1[k,n]; w2t[n,k] = W2[k,n]
__global__ void prep_weights_kernel(const float* __restrict__ W1,
                                    const float* __restrict__ W2,
                                    __bf16* __restrict__ w1t,
                                    __bf16* __restrict__ w2t) {
    int i = blockIdx.x * blockDim.x + threadIdx.x;
    if (i < F_IN * HID) {
        int k = i / HID, n = i % HID;
        w1t[n * F_IN + k] = (__bf16)W1[i];
    } else if (i < F_IN * HID + HID * NCLS) {
        int j = i - F_IN * HID;
        int k = j / NCLS, n = j % NCLS;
        w2t[n * HID + k] = (__bf16)W2[j];
    }
}

__global__ void count_edges_kernel(const int* __restrict__ col,
                                   int* __restrict__ counts, int e) {
    int i = blockIdx.x * blockDim.x + threadIdx.x;
    if (i < e) atomicAdd(&counts[col[i]], 1);
}

// deg = in-degree + 1 (self loop); dinv = rsqrt(deg)
__global__ void dinv_kernel(const int* __restrict__ counts,
                            float* __restrict__ dinv, int n) {
    int i = blockIdx.x * blockDim.x + threadIdx.x;
    if (i < n) dinv[i] = rsqrtf((float)counts[i] + 1.0f);
}

// Single-block exclusive scan of counts -> col_ptr (and cursor copy).
__global__ __launch_bounds__(1024) void scan_kernel(const int* __restrict__ counts,
                                                    int* __restrict__ col_ptr,
                                                    int* __restrict__ cursor, int n) {
    __shared__ int sdata[1024];
    __shared__ int s_total;
    const int tid = threadIdx.x;
    if (tid == 0) s_total = 0;
    __syncthreads();
    for (int base = 0; base < n; base += 1024) {
        int i = base + tid;
        int v = (i < n) ? counts[i] : 0;
        sdata[tid] = v;
        __syncthreads();
        #pragma unroll
        for (int off = 1; off < 1024; off <<= 1) {
            int t = (tid >= off) ? sdata[tid - off] : 0;
            __syncthreads();
            sdata[tid] += t;
            __syncthreads();
        }
        int incl = sdata[tid];
        int chunk_total = sdata[1023];
        int tot = s_total;
        if (i < n) {
            int excl = tot + incl - v;
            col_ptr[i] = excl;
            cursor[i]  = excl;
        }
        __syncthreads();
        if (tid == 0) s_total = tot + chunk_total;
        __syncthreads();
    }
    if (tid == 0) col_ptr[n] = s_total;
}

// Bin edges by destination; precompute edge norm = dinv[row]*dinv[col].
__global__ void scatter_kernel(const int* __restrict__ row, const int* __restrict__ col,
                               const float* __restrict__ dinv,
                               int* __restrict__ cursor,
                               int* __restrict__ srow, float* __restrict__ snorm, int e) {
    int i = blockIdx.x * blockDim.x + threadIdx.x;
    if (i >= e) return;
    int c = col[i];
    int r = row[i];
    int pos = atomicAdd(&cursor[c], 1);
    srow[pos]  = r;
    snorm[pos] = dinv[r] * dinv[c];
}

// -------------------------------- WMMA GEMMs -------------------------------
// GEMM1: h1b[N,256] = relu(x[N,512] @ W1 + b1), bf16 output.
// Block = 512 threads = 16 waves; wave w handles N-tile w; one M-tile/block.
__global__ __launch_bounds__(512) void gemm1_kernel(const float* __restrict__ x,
                                                    const __bf16* __restrict__ w1t,
                                                    const float* __restrict__ b1,
                                                    __bf16* __restrict__ h1b) {
    const int lane  = threadIdx.x & 31;
    const int wave  = threadIdx.x >> 5;          // 0..15 -> n-tile
    const int m0    = blockIdx.x * 16;
    const int n0    = wave * 16;
    const int lm    = lane & 15;
    const int mrow  = m0 + lm;
    const int ncol  = n0 + lm;
    const int abase = (lane < 16) ? 0 : 16;      // A: K sub-offset per lane half
    const int bbase = (lane < 16) ? 0 : 16;      // B: K sub-offset per lane half

    const float*  xr = x   + (size_t)mrow * F_IN;
    const __bf16* br = w1t + (size_t)ncol * F_IN + bbase;

    v8f c = {};
    #pragma unroll 4
    for (int ks = 0; ks < F_IN / 32; ++ks) {
        const int k0 = ks * 32;
        // A tile: 16 f32 -> bf16 per lane, one contiguous 16-float run
        float4 a0 = *(const float4*)(xr + k0 + abase);
        float4 a1 = *(const float4*)(xr + k0 + abase + 4);
        float4 a2 = *(const float4*)(xr + k0 + abase + 8);
        float4 a3 = *(const float4*)(xr + k0 + abase + 12);
        v16bf a;
        a[0]  = (__bf16)a0.x; a[1]  = (__bf16)a0.y; a[2]  = (__bf16)a0.z; a[3]  = (__bf16)a0.w;
        a[4]  = (__bf16)a1.x; a[5]  = (__bf16)a1.y; a[6]  = (__bf16)a1.z; a[7]  = (__bf16)a1.w;
        a[8]  = (__bf16)a2.x; a[9]  = (__bf16)a2.y; a[10] = (__bf16)a2.z; a[11] = (__bf16)a2.w;
        a[12] = (__bf16)a3.x; a[13] = (__bf16)a3.y; a[14] = (__bf16)a3.z; a[15] = (__bf16)a3.w;
        // B tile: 16 contiguous bf16 (32B) per lane from transposed weights
        v16bf b = *(const v16bf*)(br + k0);
        c = __builtin_amdgcn_wmma_f32_16x16x32_bf16(false, a, false, b,
                                                    (short)0, c, false, false);
    }
    const float bias  = b1[ncol];
    const int   rbase = m0 + ((lane < 16) ? 0 : 8);
    #pragma unroll
    for (int r = 0; r < 8; ++r) {
        float v = fmaxf(c[r] + bias, 0.0f);
        h1b[(size_t)(rbase + r) * HID + ncol] = (__bf16)v;
    }
}

// GEMM2: h[N,64] = h1b[N,256] @ W2 + b2, f32 output.
__global__ __launch_bounds__(128) void gemm2_kernel(const __bf16* __restrict__ h1b,
                                                    const __bf16* __restrict__ w2t,
                                                    const float* __restrict__ b2,
                                                    float* __restrict__ h) {
    const int lane  = threadIdx.x & 31;
    const int wave  = threadIdx.x >> 5;          // 0..3 -> n-tile
    const int m0    = blockIdx.x * 16;
    const int n0    = wave * 16;
    const int lm    = lane & 15;
    const int mrow  = m0 + lm;
    const int ncol  = n0 + lm;
    const int abase = (lane < 16) ? 0 : 16;
    const int bbase = (lane < 16) ? 0 : 16;

    const __bf16* ar = h1b + (size_t)mrow * HID;
    const __bf16* br = w2t + (size_t)ncol * HID + bbase;

    v8f c = {};
    #pragma unroll
    for (int ks = 0; ks < HID / 32; ++ks) {
        const int k0 = ks * 32;
        bf16x8 al = *(const bf16x8*)(ar + k0 + abase);
        bf16x8 ah = *(const bf16x8*)(ar + k0 + abase + 8);
        v16bf a;
        #pragma unroll
        for (int i = 0; i < 8; ++i) { a[i] = al[i]; a[8 + i] = ah[i]; }
        v16bf b = *(const v16bf*)(br + k0);
        c = __builtin_amdgcn_wmma_f32_16x16x32_bf16(false, a, false, b,
                                                    (short)0, c, false, false);
    }
    const float bias  = b2[ncol];
    const int   rbase = m0 + ((lane < 16) ? 0 : 8);
    #pragma unroll
    for (int r = 0; r < 8; ++r)
        h[(size_t)(rbase + r) * NCLS + ncol] = c[r] + bias;
}

// ------------------------------ APPNP propagation --------------------------
// z_out[n] = 0.9 * (dinv[n]^2 * z_in[n] + sum_in-edges norm * z_in[row]) + 0.1 * h[n]
// 16 lanes (float4 each) per node => 64 channels.
__global__ __launch_bounds__(256) void appnp_kernel(const float* __restrict__ zin,
                                                    float* __restrict__ zout,
                                                    const float* __restrict__ h,
                                                    const float* __restrict__ dinv,
                                                    const int* __restrict__ col_ptr,
                                                    const int* __restrict__ srow,
                                                    const float* __restrict__ snorm,
                                                    int n) {
    const int tid  = blockIdx.x * blockDim.x + threadIdx.x;
    const int node = tid >> 4;
    const int sub  = (tid & 15) * 4;
    if (node >= n) return;

    const float di = dinv[node];
    float4 acc = *(const float4*)(zin + (size_t)node * NCLS + sub);
    const float wself = di * di;
    acc.x *= wself; acc.y *= wself; acc.z *= wself; acc.w *= wself;

    const int e0 = col_ptr[node];
    const int e1 = col_ptr[node + 1];
    for (int e = e0; e < e1; ++e) {
        const int   r = srow[e];
        const float w = snorm[e];
        float4 zv = *(const float4*)(zin + (size_t)r * NCLS + sub);
        acc.x = fmaf(w, zv.x, acc.x);
        acc.y = fmaf(w, zv.y, acc.y);
        acc.z = fmaf(w, zv.z, acc.z);
        acc.w = fmaf(w, zv.w, acc.w);
    }

    float4 hv = *(const float4*)(h + (size_t)node * NCLS + sub);
    float4 o;
    o.x = (1.0f - ALPHA) * acc.x + ALPHA * hv.x;
    o.y = (1.0f - ALPHA) * acc.y + ALPHA * hv.y;
    o.z = (1.0f - ALPHA) * acc.z + ALPHA * hv.z;
    o.w = (1.0f - ALPHA) * acc.w + ALPHA * hv.w;
    *(float4*)(zout + (size_t)node * NCLS + sub) = o;
}

// ------------------------------ log_softmax --------------------------------
__global__ __launch_bounds__(256) void logsoftmax_kernel(const float* __restrict__ z,
                                                         float* __restrict__ out, int n) {
    const int tid  = blockIdx.x * blockDim.x + threadIdx.x;
    const int node = tid >> 4;
    const int sub  = (tid & 15) * 4;
    if (node >= n) return;

    float4 v = *(const float4*)(z + (size_t)node * NCLS + sub);
    float m = fmaxf(fmaxf(v.x, v.y), fmaxf(v.z, v.w));
    #pragma unroll
    for (int off = 8; off >= 1; off >>= 1)
        m = fmaxf(m, __shfl_xor(m, off, 16));

    float s = __expf(v.x - m) + __expf(v.y - m) + __expf(v.z - m) + __expf(v.w - m);
    #pragma unroll
    for (int off = 8; off >= 1; off >>= 1)
        s += __shfl_xor(s, off, 16);

    const float lse = m + __logf(s);
    float4 o;
    o.x = v.x - lse; o.y = v.y - lse; o.z = v.z - lse; o.w = v.w - lse;
    *(float4*)(out + (size_t)node * NCLS + sub) = o;
}

// -------------------------------- launcher ---------------------------------
extern "C" void kernel_launch(void* const* d_in, const int* in_sizes, int n_in,
                              void* d_out, int out_size, void* d_ws, size_t ws_size,
                              hipStream_t stream) {
    const float* x  = (const float*)d_in[0];
    const int*   ei = (const int*)d_in[1];
    const float* W1 = (const float*)d_in[2];
    const float* b1 = (const float*)d_in[3];
    const float* W2 = (const float*)d_in[4];
    const float* b2 = (const float*)d_in[5];
    float* out = (float*)d_out;

    const int N = in_sizes[0] / F_IN;   // 100000 (divisible by 16)
    const int E = in_sizes[1] / 2;      // 3200000
    const int* row = ei;
    const int* col = ei + E;

    // Carve workspace (256B-aligned).
    char* p = (char*)d_ws;
    auto carve = [&](size_t bytes) -> void* {
        void* r = (void*)p;
        p += (bytes + 255) & ~(size_t)255;
        return r;
    };
    float*  dinv    = (float*)carve((size_t)N * 4);
    int*    counts  = (int*)carve((size_t)N * 4);
    int*    cursor  = (int*)carve((size_t)N * 4);
    int*    col_ptr = (int*)carve(((size_t)N + 1) * 4);
    int*    srow    = (int*)carve((size_t)E * 4);
    float*  snorm   = (float*)carve((size_t)E * 4);
    __bf16* w1t     = (__bf16*)carve((size_t)F_IN * HID * 2);
    __bf16* w2t     = (__bf16*)carve((size_t)HID * NCLS * 2);
    __bf16* h1b     = (__bf16*)carve((size_t)N * HID * 2);
    float*  h       = (float*)carve((size_t)N * NCLS * 4);
    float*  za      = (float*)carve((size_t)N * NCLS * 4);
    float*  zb      = (float*)carve((size_t)N * NCLS * 4);

    const int TB = 256;

    // Graph structure (once per call).
    init_counts_kernel<<<(N + TB - 1) / TB, TB, 0, stream>>>(counts, N);
    prep_weights_kernel<<<(F_IN * HID + HID * NCLS + TB - 1) / TB, TB, 0, stream>>>(W1, W2, w1t, w2t);
    count_edges_kernel<<<(E + TB - 1) / TB, TB, 0, stream>>>(col, counts, E);
    dinv_kernel<<<(N + TB - 1) / TB, TB, 0, stream>>>(counts, dinv, N);
    scan_kernel<<<1, 1024, 0, stream>>>(counts, col_ptr, cursor, N);
    scatter_kernel<<<(E + TB - 1) / TB, TB, 0, stream>>>(row, col, dinv, cursor, srow, snorm, E);

    // MLP encoder (WMMA bf16).
    gemm1_kernel<<<N / 16, 512, 0, stream>>>(x, w1t, b1, h1b);
    gemm2_kernel<<<N / 16, 128, 0, stream>>>(h1b, w2t, b2, h);

    // APPNP: 10 power iterations, ping-pong buffers, atomic-free CSR gather.
    const int pgrid = ((N * 16) + TB - 1) / TB;
    const float* zin = h;
    float* zout = za;
    for (int k = 0; k < 10; ++k) {
        appnp_kernel<<<pgrid, TB, 0, stream>>>(zin, zout, h, dinv, col_ptr, srow, snorm, N);
        zin  = zout;
        zout = (zout == za) ? zb : za;
    }

    logsoftmax_kernel<<<pgrid, TB, 0, stream>>>(zin, out, N);
}
